// MultiHeadAttention_7670811591338
// MI455X (gfx1250) — matrix operand order repaired and checked by reading, not compiled, not run on hardware
//
#include <hip/hip_runtime.h>
#include <hip/hip_bf16.h>
#include <math.h>

// ---------------------------------------------------------------------------
// MI455X (gfx1250) fused MHA block: QKV proj -> flash attention -> out proj +
// residual + LayerNorm.  All GEMMs: v_wmma_f32_16x16x32_bf16 (wave32).
// LDS tiles laid out so every WMMA fragment is 2x ds_load_b128.
// bf16->LDS copies that need no convert/transpose use the CDNA5 async path
// (global_load_async_to_lds_b128 + ASYNCcnt).
// Intermediates (Q,K,V,attn-out) are bf16 in d_ws (64 MB total -> fits L2).
// ---------------------------------------------------------------------------

typedef __bf16 bf16_t;
typedef __attribute__((ext_vector_type(16))) __bf16 v16bf;
typedef __attribute__((ext_vector_type(8)))  __bf16 v8bf;
typedef __attribute__((ext_vector_type(8)))  float  v8f;

#define BS_    4
#define SEQ_   1024
#define NDIM_  128
#define NHEAD_ 16
#define INNER_ 2048
#define ROWS_  (BS_ * SEQ_)       /* 4096 */
#define QKVCOLS_ (3 * INNER_)     /* 6144 */

// ---- WMMA helpers ---------------------------------------------------------

__device__ __forceinline__ v8f wmma_bf16(v16bf a, v16bf b, v8f c) {
  return __builtin_amdgcn_wmma_f32_16x16x32_bf16(false, a, false, b, (short)0,
                                                 c, false, false);
}

__device__ __forceinline__ v16bf cat8(v8bf lo, v8bf hi) {
  return __builtin_shufflevector(lo, hi, 0, 1, 2, 3, 4, 5, 6, 7, 8, 9, 10, 11,
                                 12, 13, 14, 15);
}

// A fragment: 16x32, src row-major [M][K]. Lane(m,half) reads K = half*8..+7
// and 16+half*8..+7 -> two aligned 16B loads.
__device__ __forceinline__ v16bf load_a_frag(const bf16_t* src, int ld,
                                             int lane) {
  const int half = lane >> 4, m = lane & 15;
  const bf16_t* p = src + m * ld + half * 8;
  const v8bf lo = *(const v8bf*)p;
  const v8bf hi = *(const v8bf*)(p + 16);
  return cat8(lo, hi);
}

// B fragment: 32x16 stored TRANSPOSED [N][K]. Lane(n,half) reads K =
// half*8..+7 and 16+half*8..+7 of column n (2x 16B loads).
__device__ __forceinline__ v16bf load_bt_frag(const bf16_t* src, int ld,
                                              int lane) {
  const int half = lane >> 4, n = lane & 15;
  const bf16_t* p = src + n * ld + half * 8;
  const v8bf lo = *(const v8bf*)p;
  const v8bf hi = *(const v8bf*)(p + 16);
  return cat8(lo, hi);
}

__device__ __forceinline__ v8f v8f_zero() {
  v8f z;
#pragma unroll
  for (int i = 0; i < 8; ++i) z[i] = 0.0f;
  return z;
}

__device__ __forceinline__ bf16_t u16_as_bf16(unsigned short u) {
  return __builtin_bit_cast(bf16_t, u);
}

// ---- CDNA5 async Global->LDS copy (no VGPR round-trip, ASYNCcnt) ----------
// GV mode: VDST = LDS byte offset (low 32 bits of flat LDS address),
// VADDR = 64-bit global address, SADDR = off.
__device__ __forceinline__ void async_copy_b128(void* lds_dst,
                                                const void* gsrc) {
  const unsigned lds_off = (unsigned)(unsigned long long)lds_dst;
  asm volatile("global_load_async_to_lds_b128 %0, %1, off"
               :: "v"(lds_off), "v"(gsrc)
               : "memory");
}

__device__ __forceinline__ void wait_asynccnt0() {
#if __has_builtin(__builtin_amdgcn_s_wait_asynccnt)
  __builtin_amdgcn_s_wait_asynccnt(0);
#else
  asm volatile("s_wait_asynccnt 0" ::: "memory");
#endif
}

// ---------------------------------------------------------------------------
// Kernel 1: qkv = X @ Wqkv + bqkv ; scatter bf16 into Q/K/V [b,h,s,d]
// Grid: 32 row-blocks x 48 col-blocks; 256 thr (8 waves).
// Xs (A-layout) and the transposed W stage buffer alias the same LDS.
// ---------------------------------------------------------------------------
__global__ __launch_bounds__(256) void qkv_proj_kernel(
    const float* __restrict__ X, const float* __restrict__ Wqkv,
    const float* __restrict__ bqkv, bf16_t* __restrict__ Qw,
    bf16_t* __restrict__ Kw, bf16_t* __restrict__ Vw) {
  __shared__ __align__(16) unsigned char smem[128 * 152 * 2];  // 38912 B
  bf16_t(*Xs)[152] = (bf16_t(*)[152])smem;  // [M=128][K=128 pad24], 304B rows
  bf16_t(*Wt)[72] = (bf16_t(*)[72])smem;    // [N=128][K=64 pad8], 144B rows
  const int tid = threadIdx.x, lane = tid & 31, wave = tid >> 5;
  const int row0 = (blockIdx.x & 31) * 128;
  const int col0 = (blockIdx.x >> 5) * 128;

  // stage X tile (fp32 -> bf16), row-major
  for (int idx = tid * 4; idx < 128 * 128; idx += 256 * 4) {
    const int r = idx >> 7, c = idx & 127;
    const float4 f = *(const float4*)(X + (size_t)(row0 + r) * NDIM_ + c);
    Xs[r][c] = (bf16_t)f.x; Xs[r][c + 1] = (bf16_t)f.y;
    Xs[r][c + 2] = (bf16_t)f.z; Xs[r][c + 3] = (bf16_t)f.w;
  }
  __syncthreads();

  v16bf afrag[4];
#pragma unroll
  for (int kk = 0; kk < 4; ++kk)
    afrag[kk] = load_a_frag(&Xs[wave * 16][kk * 32], 152, lane);
  __syncthreads();  // everyone done reading Xs before Wt overwrites it

  v8f acc[8];
#pragma unroll
  for (int j = 0; j < 8; ++j) acc[j] = v8f_zero();

  for (int ks = 0; ks < 2; ++ks) {
    // stage W K-half, transposed [col][k]
    for (int idx = tid * 4; idx < 64 * 128; idx += 256 * 4) {
      const int r = idx >> 7, c = idx & 127;  // r = K row, c = out col
      const float4 f =
          *(const float4*)(Wqkv + (size_t)(ks * 64 + r) * QKVCOLS_ + col0 + c);
      Wt[c][r] = (bf16_t)f.x; Wt[c + 1][r] = (bf16_t)f.y;
      Wt[c + 2][r] = (bf16_t)f.z; Wt[c + 3][r] = (bf16_t)f.w;
    }
    __syncthreads();
#pragma unroll
    for (int j = 0; j < 8; ++j) {
#pragma unroll
      for (int kk = 0; kk < 2; ++kk) {
        const v16bf b = load_bt_frag(&Wt[j * 16][kk * 32], 72, lane);
        acc[j] = wmma_bf16(afrag[ks * 2 + kk], b, acc[j]);
      }
    }
    __syncthreads();
  }

  // epilogue: + bias, scatter to Q/K/V in [b,h,s,d]
  const int half = lane >> 4, n = lane & 15;
#pragma unroll
  for (int j = 0; j < 8; ++j) {
    const int gcol = col0 + j * 16 + n;
    const int which = gcol >> 11;  // 0=Q 1=K 2=V
    const int rem = gcol & 2047;
    const int h = rem >> 7, d = rem & 127;
    const float bias = bqkv[gcol];
    bf16_t* dst = (which == 0) ? Qw : ((which == 1) ? Kw : Vw);
#pragma unroll
    for (int r = 0; r < 8; ++r) {
      const int grow = row0 + wave * 16 + r + 8 * half;
      const int b_ = grow >> 10, s = grow & 1023;
      dst[(((size_t)(b_ * NHEAD_ + h) * SEQ_) + s) * NDIM_ + d] =
          (bf16_t)(acc[j][r] + bias);
    }
  }
}

// ---------------------------------------------------------------------------
// Kernel 2: flash attention per (b,h). Grid: 64 bh x 8 q-tiles of 128.
// K staged with async Global->LDS (row-major = B^T layout for Q@K^T);
// V staged transposed through VGPRs.
// ---------------------------------------------------------------------------
__global__ __launch_bounds__(256) void attn_kernel(
    const bf16_t* __restrict__ Qw, const bf16_t* __restrict__ Kw,
    const bf16_t* __restrict__ Vw, bf16_t* __restrict__ AO) {
  __shared__ __align__(16) bf16_t Ks[32][152];    // 304B rows, conflict-free
  __shared__ __align__(16) bf16_t Vt[128][40];    // 80B rows, conflict-free
  __shared__ __align__(16) bf16_t Ps[8][16][56];  // per-wave P, 112B rows
  const int tid = threadIdx.x, lane = tid & 31, wave = tid >> 5;
  const int bh = blockIdx.x >> 3;
  const int qt = blockIdx.x & 7;
  const bf16_t* Qb = Qw + (size_t)bh * SEQ_ * NDIM_;
  const bf16_t* Kb = Kw + (size_t)bh * SEQ_ * NDIM_;
  const bf16_t* Vb = Vw + (size_t)bh * SEQ_ * NDIM_;
  const int q0 = qt * 128 + wave * 16;

  v16bf qfrag[4];
#pragma unroll
  for (int kk = 0; kk < 4; ++kk)  // global_load_b128 x2 per fragment
    qfrag[kk] = load_a_frag(Qb + (size_t)q0 * NDIM_ + kk * 32, NDIM_, lane);

  v8f acc[8];
#pragma unroll
  for (int j = 0; j < 8; ++j) acc[j] = v8f_zero();
  float mrow[8], lrow[8];
#pragma unroll
  for (int r = 0; r < 8; ++r) { mrow[r] = -1e30f; lrow[r] = 0.0f; }

  const float scale = 11.313708498984761f;  // sqrt(128)

  for (int kb = 0; kb < SEQ_ / 32; ++kb) {
    __syncthreads();
    {
      // K tile: async copy, 16B chunks; LDS row padding handled per-chunk
      const bf16_t* kblk = Kb + (size_t)kb * 32 * NDIM_;
#pragma unroll
      for (int v = tid; v < 512; v += 256) {   // 32 rows x 16 chunks
        const int r = v >> 4, c = v & 15;
        async_copy_b128(&Ks[r][c * 8], kblk + r * NDIM_ + c * 8);
      }
      // V tile: transpose through VGPRs (needed for PV B^T layout)
      const unsigned int* vsrc =
          (const unsigned int*)(Vb + (size_t)kb * 32 * NDIM_);
      for (int v = tid; v < 32 * 64; v += 256) {
        const int r = v >> 6, c = v & 63;
        const unsigned int w = vsrc[r * 64 + c];
        Vt[c * 2][r] = u16_as_bf16((unsigned short)(w & 0xffffu));
        Vt[c * 2 + 1][r] = u16_as_bf16((unsigned short)(w >> 16));
      }
      if (kb + 1 < SEQ_ / 32) {  // global_prefetch_b8 for next V block
        __builtin_prefetch(Vb + (size_t)(kb + 1) * 32 * NDIM_ + tid * 32, 0, 0);
      }
    }
    wait_asynccnt0();
    __syncthreads();

    // S = Q @ K^T  (two 16-key tiles)
    v8f s0 = v8f_zero(), s1 = v8f_zero();
#pragma unroll
    for (int kk = 0; kk < 4; ++kk) {
      s0 = wmma_bf16(qfrag[kk], load_bt_frag(&Ks[0][kk * 32], 152, lane), s0);
      s1 = wmma_bf16(qfrag[kk], load_bt_frag(&Ks[16][kk * 32], 152, lane), s1);
    }

    float mcur[8];
#pragma unroll
    for (int r = 0; r < 8; ++r) {
      s0[r] *= scale; s1[r] *= scale;
      mcur[r] = fmaxf(s0[r], s1[r]);
    }
#pragma unroll
    for (int msk = 1; msk < 16; msk <<= 1) {
#pragma unroll
      for (int r = 0; r < 8; ++r)
        mcur[r] = fmaxf(mcur[r], __shfl_xor(mcur[r], msk, 32));
    }

    float corr[8], psum[8];
#pragma unroll
    for (int r = 0; r < 8; ++r) {
      const float mn = fmaxf(mrow[r], mcur[r]);
      corr[r] = __expf(mrow[r] - mn);
      mrow[r] = mn;
      const float p0 = __expf(s0[r] - mn);
      const float p1 = __expf(s1[r] - mn);
      s0[r] = p0; s1[r] = p1;
      psum[r] = p0 + p1;
    }
#pragma unroll
    for (int msk = 1; msk < 16; msk <<= 1) {
#pragma unroll
      for (int r = 0; r < 8; ++r) psum[r] += __shfl_xor(psum[r], msk, 32);
    }
#pragma unroll
    for (int r = 0; r < 8; ++r) lrow[r] = lrow[r] * corr[r] + psum[r];

    // P (C-layout fp32) -> per-wave LDS bounce -> A-layout bf16 fragment
    const int half = lane >> 4, n = lane & 15;
#pragma unroll
    for (int r = 0; r < 8; ++r) {
      Ps[wave][r + 8 * half][n] = (bf16_t)s0[r];
      Ps[wave][r + 8 * half][16 + n] = (bf16_t)s1[r];
    }
    asm volatile("s_wait_dscnt 0" ::: "memory");  // wave-private RAW fence
    const v16bf pfrag = load_a_frag(&Ps[wave][0][0], 56, lane);

    // O += P @ V  (8 d-tiles), online accumulator rescale
#pragma unroll
    for (int dt = 0; dt < 8; ++dt) {
#pragma unroll
      for (int r = 0; r < 8; ++r) acc[dt][r] *= corr[r];
      acc[dt] =
          wmma_bf16(pfrag, load_bt_frag(&Vt[dt * 16][0], 40, lane), acc[dt]);
    }
  }

  // epilogue: normalize by l, write bf16 attn-out [b, s, h*128+d]
  const int half = lane >> 4, n = lane & 15;
  const int b_ = bh >> 4, h = bh & 15;
#pragma unroll
  for (int dt = 0; dt < 8; ++dt) {
#pragma unroll
    for (int r = 0; r < 8; ++r) {
      const int s = qt * 128 + wave * 16 + r + 8 * half;
      AO[((size_t)(b_ * SEQ_ + s)) * INNER_ + h * NDIM_ + dt * 16 + n] =
          (bf16_t)(acc[dt][r] / lrow[r]);
    }
  }
}

// ---------------------------------------------------------------------------
// Kernel 3: out = LN(X + AO @ Wo + bo). Grid: 32 row-blocks of 128.
// A tiles staged with async Global->LDS; Wo tiles staged transposed.
// LN fully in-register via 16-lane shfl.
// ---------------------------------------------------------------------------
__global__ __launch_bounds__(256) void oproj_ln_kernel(
    const bf16_t* __restrict__ AO, const float* __restrict__ Wo,
    const float* __restrict__ bo, const float* __restrict__ X,
    const float* __restrict__ lns, const float* __restrict__ lnb,
    float* __restrict__ out) {
  __shared__ __align__(16) bf16_t As[128][40];  // [row][k32 pad8], 80B rows
  __shared__ __align__(16) bf16_t Bt[128][40];  // Wo^T [col][k32 pad8]
  const int tid = threadIdx.x, lane = tid & 31, wave = tid >> 5;
  const int row0 = blockIdx.x * 128;

  v8f acc[8];
#pragma unroll
  for (int j = 0; j < 8; ++j) acc[j] = v8f_zero();

  for (int kb = 0; kb < INNER_ / 32; ++kb) {
    __syncthreads();
    // A tile: async copy, 128 rows x 4 chunks of 16B (64B data + 16B pad row)
#pragma unroll
    for (int v = tid; v < 512; v += 256) {
      const int r = v >> 2, c = v & 3;
      async_copy_b128(&As[r][c * 8],
                      AO + (size_t)(row0 + r) * INNER_ + kb * 32 + c * 8);
    }
    for (int idx = tid * 4; idx < 32 * 128; idx += 256 * 4) {  // Wo^T tile
      const int r = idx >> 7, c = idx & 127;
      const float4 f = *(const float4*)(Wo + (size_t)(kb * 32 + r) * NDIM_ + c);
      Bt[c][r] = (bf16_t)f.x; Bt[c + 1][r] = (bf16_t)f.y;
      Bt[c + 2][r] = (bf16_t)f.z; Bt[c + 3][r] = (bf16_t)f.w;
    }
    wait_asynccnt0();
    __syncthreads();
    const v16bf a = load_a_frag(&As[wave * 16][0], 40, lane);
#pragma unroll
    for (int j = 0; j < 8; ++j)
      acc[j] = wmma_bf16(a, load_bt_frag(&Bt[j * 16][0], 40, lane), acc[j]);
  }

  // fused epilogue: + bo + residual, LayerNorm per row (eps 1e-6)
  const int half = lane >> 4, n = lane & 15;
  float bov[8], lsv[8], lbv[8];
#pragma unroll
  for (int j = 0; j < 8; ++j) {
    bov[j] = bo[j * 16 + n];
    lsv[j] = lns[j * 16 + n];
    lbv[j] = lnb[j * 16 + n];
  }
#pragma unroll
  for (int r = 0; r < 8; ++r) {
    const int row = row0 + wave * 16 + r + 8 * half;
    float vals[8], sum = 0.0f, sq = 0.0f;
#pragma unroll
    for (int j = 0; j < 8; ++j) {
      const float v = acc[j][r] + bov[j] + X[(size_t)row * NDIM_ + j * 16 + n];
      vals[j] = v;
      sum += v;
      sq += v * v;
    }
#pragma unroll
    for (int msk = 1; msk < 16; msk <<= 1) {  // whole row lives in 16 lanes
      sum += __shfl_xor(sum, msk, 32);
      sq += __shfl_xor(sq, msk, 32);
    }
    const float mean = sum * (1.0f / 128.0f);
    const float var = sq * (1.0f / 128.0f) - mean * mean;
    const float rstd = rsqrtf(var + 1e-6f);
#pragma unroll
    for (int j = 0; j < 8; ++j)
      out[(size_t)row * NDIM_ + j * 16 + n] =
          (vals[j] - mean) * rstd * lsv[j] + lbv[j];
  }
}

// ---------------------------------------------------------------------------
extern "C" void kernel_launch(void* const* d_in, const int* in_sizes, int n_in,
                              void* d_out, int out_size, void* d_ws,
                              size_t ws_size, hipStream_t stream) {
  (void)in_sizes; (void)n_in; (void)out_size; (void)ws_size;
  const float* X    = (const float*)d_in[0];
  const float* Wqkv = (const float*)d_in[1];
  const float* bqkv = (const float*)d_in[2];
  const float* Wo   = (const float*)d_in[3];
  const float* bo   = (const float*)d_in[4];
  const float* lns  = (const float*)d_in[5];
  const float* lnb  = (const float*)d_in[6];
  float* out = (float*)d_out;

  const size_t NQ = (size_t)ROWS_ * INNER_;  // 8,388,608 elems / tensor
  bf16_t* Qw = (bf16_t*)d_ws;                // 16 MB
  bf16_t* Kw = Qw + NQ;                      // 16 MB
  bf16_t* Vw = Kw + NQ;                      // 16 MB
  bf16_t* AO = Vw + NQ;                      // 16 MB  (64 MB total, fits L2)

  qkv_proj_kernel<<<dim3(32 * 48), 256, 0, stream>>>(X, Wqkv, bqkv, Qw, Kw, Vw);
  attn_kernel<<<dim3(64 * 8), 256, 0, stream>>>(Qw, Kw, Vw, AO);
  oproj_ln_kernel<<<dim3(ROWS_ / 128), 256, 0, stream>>>(AO, Wo, bo, X, lns,
                                                         lnb, out);
}
